// SRNN_multi_14989435863737
// MI455X (gfx1250) — matrix operand order repaired and checked by reading, not compiled, not run on hardware
//
#include <hip/hip_runtime.h>
#include <hip/hip_bf16.h>

// SRNN forward on gfx1250 (MI455X), wave32 + WMMA bf16.
//
// Shapes: N=512, T=1000, D_IN=HID=128, D_OUT=20.
// Grid: 32 blocks (one 16-row batch tile each). Block: 256 threads = 8 waves.
// Wave j owns hidden columns [16j,16j+16). All weights are held in VGPRs as
// persistent WMMA B-fragments; spikes/x go through LDS in bf16.

#define N_BATCH 512
#define T_STEPS 1000
#define DIM     128     // D_IN == HID
#define D_OUT   20

typedef __attribute__((ext_vector_type(16))) __bf16 v16bf;
typedef __attribute__((ext_vector_type(8)))  __bf16 v8bf;
typedef __attribute__((ext_vector_type(8)))  float  v8f;
typedef __attribute__((ext_vector_type(4)))  float  v4f;

__device__ __forceinline__ v8f wmma_bf16(v16bf a, v16bf b, v8f c) {
    // (neg_a, A, neg_b, B, c_mod, C, reuse_a, reuse_b)
    return __builtin_amdgcn_wmma_f32_16x16x32_bf16(
        false, a, false, b, (short)0, c, false, false);
}

// Build one B-fragment (32x16 bf16, K-tile kt) for this lane from a row-major
// fp32 weight matrix W[out][128]; B[k][n] = W[n][k].
// Lane layout (dense 32x16, 16-bit): elements 0-7 hold K=8*hi+(0..7),
// elements 8-15 hold K=16+8*hi+(0..7) (col = lane%16, hi = lane/16).
__device__ __forceinline__ v16bf make_bfrag(const float* W, int col, int kt,
                                            int hi, bool valid) {
    v16bf b;
    if (valid) {
        const float* p = W + (size_t)col * DIM + kt * 32 + hi * 8;
#pragma unroll
        for (int i = 0; i < 16; ++i) b[i] = (__bf16)p[i < 8 ? i : i + 8];
    } else {
#pragma unroll
        for (int i = 0; i < 16; ++i) b[i] = (__bf16)0.0f;
    }
    return b;
}

// Load one A-fragment (16x32 bf16, K-tile kt) from a row-major [16][128] bf16
// LDS buffer. Lane L: M = L%16, hi = L/16.
//   VGPRs 0-3: K = kt*32 + 8*hi + (0..7)   -> one 16B ds_load
//   VGPRs 4-7: K = kt*32 + 16 + 8*hi + (0..7) -> one 16B ds_load
__device__ __forceinline__ v16bf load_afrag(const unsigned short* buf, int m,
                                            int hi, int kt) {
    union { v16bf v; v8bf h[2]; } u;
    const unsigned short* base = buf + m * DIM + kt * 32 + hi * 8;
    u.h[0] = *(const v8bf*)(base);
    u.h[1] = *(const v8bf*)(base + 16);
    return u.v;
}

__global__ __launch_bounds__(256)
void srnn_kernel(const float* __restrict__ x,
                 const float* __restrict__ Wi1, const float* __restrict__ bi1,
                 const float* __restrict__ Wh1, const float* __restrict__ bh1,
                 const float* __restrict__ Wi2, const float* __restrict__ bi2,
                 const float* __restrict__ Wh2, const float* __restrict__ bh2,
                 const float* __restrict__ Wo,  const float* __restrict__ bo,
                 float* __restrict__ out) {
    __shared__ __align__(16) unsigned short xbuf[16 * DIM];
    __shared__ __align__(16) unsigned short s1buf[2][16 * DIM];
    __shared__ __align__(16) unsigned short s2buf[2][16 * DIM];

    const int tid  = threadIdx.x;
    const int wave = tid >> 5;     // 0..7 (wave32)
    const int lane = tid & 31;
    const int nloc = lane & 15;    // column within 16-wide slice / A-row M
    const int hi   = lane >> 4;    // K-half selector
    const int n0   = blockIdx.x * 16;       // batch row tile base
    const int col  = wave * 16 + nloc;      // hidden column owned by lane

    // ---- persistent weight B-fragments (VGPR-resident for the whole T loop)
    v16bf wi1f[4], wh1f[4], wi2f[4], wh2f[4], wof[4];
    const bool ocol_valid = (wave < 2) && (col < D_OUT);
#pragma unroll
    for (int kt = 0; kt < 4; ++kt) {
        wi1f[kt] = make_bfrag(Wi1, col, kt, hi, true);
        wh1f[kt] = make_bfrag(Wh1, col, kt, hi, true);
        wi2f[kt] = make_bfrag(Wi2, col, kt, hi, true);
        wh2f[kt] = make_bfrag(Wh2, col, kt, hi, true);
        wof[kt]  = make_bfrag(Wo,  col, kt, hi, ocol_valid);
    }
    const float bias1 = bi1[col] + bh1[col];
    const float bias2 = bi2[col] + bh2[col];
    const float bo_l  = ocol_valid ? bo[col] : 0.0f;

    // zero initial-state buffers (parity 1 is read at t=0)
    for (int i = tid; i < 16 * DIM; i += 256) {
        s1buf[1][i] = 0; s2buf[1][i] = 0;
    }

    v8f oacc = {0.f, 0.f, 0.f, 0.f, 0.f, 0.f, 0.f, 0.f};

    // coop x-tile loader indices: 8 consecutive floats per thread
    const int xr = tid >> 4;
    const int xc = (tid & 15) << 3;

    for (int t = 0; t < T_STEPS; ++t) {
        // ---- cooperative load of x[:, t, :] tile -> bf16 LDS
        {
            const float* src =
                x + ((size_t)(n0 + xr) * T_STEPS + (size_t)t) * DIM + xc;
            v4f f0 = *(const v4f*)(src);
            v4f f1 = *(const v4f*)(src + 4);
            v8bf b;
#pragma unroll
            for (int i = 0; i < 4; ++i) {
                b[i]     = (__bf16)f0[i];
                b[i + 4] = (__bf16)f1[i];
            }
            *(v8bf*)(xbuf + xr * DIM + xc) = b;
        }
        __syncthreads();   // x tile ready; prior-step reads of s1 old buf done

        const unsigned short* s1r = s1buf[(t + 1) & 1];
        unsigned short*       s1w = s1buf[t & 1];
        const unsigned short* s2r = s2buf[(t + 1) & 1];
        unsigned short*       s2w = s2buf[t & 1];

        // ---- layer 1: h1 = x@Wi1^T + s1_{t-1}@Wh1^T (+bias); spike
        {
            v8f a0 = {0.f, 0.f, 0.f, 0.f, 0.f, 0.f, 0.f, 0.f};
            v8f a1 = {0.f, 0.f, 0.f, 0.f, 0.f, 0.f, 0.f, 0.f};
#pragma unroll
            for (int kt = 0; kt < 4; ++kt) {
                v16bf xa = load_afrag(xbuf, nloc, hi, kt);
                a0 = wmma_bf16(xa, wi1f[kt], a0);
                v16bf sa = load_afrag(s1r, nloc, hi, kt);
                a1 = wmma_bf16(sa, wh1f[kt], a1);
            }
#pragma unroll
            for (int r = 0; r < 8; ++r) {
                float h = a0[r] + a1[r] + bias1;
                // D layout: row = r + 8*hi, col = this lane's column
                s1w[(r + hi * 8) * DIM + col] =
                    (h >= 1.0f) ? (unsigned short)0x3F80 : (unsigned short)0;
            }
        }
        __syncthreads();   // s1_t visible to all waves

        // ---- layer 2: h2 = s1_t@Wi2^T + s2_{t-1}@Wh2^T (+bias); spike
        {
            v8f a0 = {0.f, 0.f, 0.f, 0.f, 0.f, 0.f, 0.f, 0.f};
            v8f a1 = {0.f, 0.f, 0.f, 0.f, 0.f, 0.f, 0.f, 0.f};
#pragma unroll
            for (int kt = 0; kt < 4; ++kt) {
                v16bf sa = load_afrag(s1w, nloc, hi, kt);
                a0 = wmma_bf16(sa, wi2f[kt], a0);
                v16bf za = load_afrag(s2r, nloc, hi, kt);
                a1 = wmma_bf16(za, wh2f[kt], a1);
            }
#pragma unroll
            for (int r = 0; r < 8; ++r) {
                float h = a0[r] + a1[r] + bias2;
                s2w[(r + hi * 8) * DIM + col] =
                    (h >= 1.0f) ? (unsigned short)0x3F80 : (unsigned short)0;
            }
        }
        __syncthreads();   // s2_t visible

        // ---- readout: oacc += s2_t @ Wo^T  (accumulates in WMMA C regs)
        if (wave < 2) {
#pragma unroll
            for (int kt = 0; kt < 4; ++kt) {
                v16bf sa = load_afrag(s2w, nloc, hi, kt);
                oacc = wmma_bf16(sa, wof[kt], oacc);
            }
        }
    }

    // ---- epilogue: out = oacc/T + bo
    if (ocol_valid) {
#pragma unroll
        for (int r = 0; r < 8; ++r) {
            int row = n0 + r + hi * 8;
            out[row * D_OUT + col] = oacc[r] * (1.0f / (float)T_STEPS) + bo_l;
        }
    }
}

extern "C" void kernel_launch(void* const* d_in, const int* in_sizes, int n_in,
                              void* d_out, int out_size, void* d_ws, size_t ws_size,
                              hipStream_t stream) {
    const float* x   = (const float*)d_in[0];
    const float* Wi1 = (const float*)d_in[1];
    const float* bi1 = (const float*)d_in[2];
    const float* Wh1 = (const float*)d_in[3];
    const float* bh1 = (const float*)d_in[4];
    const float* Wi2 = (const float*)d_in[5];
    const float* bi2 = (const float*)d_in[6];
    const float* Wh2 = (const float*)d_in[7];
    const float* bh2 = (const float*)d_in[8];
    const float* Wo  = (const float*)d_in[9];
    const float* bo  = (const float*)d_in[10];
    float* out = (float*)d_out;

    dim3 grid(N_BATCH / 16);   // 32 blocks
    dim3 block(256);           // 8 wave32 waves
    srnn_kernel<<<grid, block, 0, stream>>>(x, Wi1, bi1, Wh1, bh1,
                                            Wi2, bi2, Wh2, bh2, Wo, bo, out);
}
